// GraphLearn_81776177316090
// MI455X (gfx1250) — matrix operand order repaired and checked by reading, not compiled, not run on hardware
//
#include <hip/hip_runtime.h>
#include <hip/hip_fp16.h>

// Problem constants from the reference: x[N,V,F], a[F,1], alpha scalar.
#define NB 4
#define V  512
#define F  128
#define VV (V * V)

typedef __attribute__((ext_vector_type(16))) _Float16 v16h;
typedef __attribute__((ext_vector_type(8)))  _Float16 v8h;
typedef __attribute__((ext_vector_type(8)))  float    v8f;

// ---------------------------------------------------------------------------
// Kernel 1: convert x to f16 (for WMMA A/B operands) and compute per-row
// squared norms n[b,i] = sum_f x^2 (for the L2 Gram-matrix identity).
// One 128-thread block per row.
// ---------------------------------------------------------------------------
__global__ void prep_kernel(const float* __restrict__ x,
                            _Float16* __restrict__ xh,
                            float* __restrict__ nsq) {
    const int row = blockIdx.x;          // 0 .. NB*V-1
    const int f   = threadIdx.x;         // 0 .. F-1
    const float v = x[(size_t)row * F + f];
    xh[(size_t)row * F + f] = (_Float16)v;
    __shared__ float red[F];
    red[f] = v * v;
    __syncthreads();
    for (int s = F / 2; s > 0; s >>= 1) {
        if (f < s) red[f] += red[f + s];
        __syncthreads();
    }
    if (f == 0) nsq[row] = red[0];
}

// ---------------------------------------------------------------------------
// Kernel 2: one wave32 per 16x16 (i,j) tile of one batch.
//  - Gram tile G = Xi * Xj^T via 4x v_wmma_f32_16x16x32_f16 (K = 128)
//  - weighted-L1 tile from f32 LDS copies (exact path for the softmax)
// Emits E = exp(-relu(L1)) and D2 = n_i + n_j - 2*G.
//
// ISA 7.12.2 operand layouts (wave32):
//  A 16x32 f16 : lane L -> M = L%16 ; half v -> K = (v/8)*16 + (L/16)*8 + v%8
//  B 32x16 f16 : lane L -> N = L%16 ; half v -> K as for A (code: K = (L/16)*16 + v, A and B)
//  D 16x16 f32 : lane L, reg r -> M = (L/16)*8 + r , N = L%16
// ---------------------------------------------------------------------------
__global__ void __launch_bounds__(32)
tile_kernel(const float* __restrict__ x,
            const _Float16* __restrict__ xh,
            const float* __restrict__ aw,
            const float* __restrict__ nsq,
            float* __restrict__ E,
            float* __restrict__ D2) {
    const int blk  = blockIdx.x;         // NB * 32 * 32 tiles
    const int b    = blk >> 10;
    const int t    = blk & 1023;
    const int ti   = t >> 5;
    const int tj   = t & 31;
    const int lane = threadIdx.x;
    const int half = lane >> 4;          // half-wave id
    const int lm   = lane & 15;

    // f32 tiles in LDS, rows padded to 132 floats (132%64==4) so the 16
    // distinct-row xj reads land in distinct banks; xi reads are half-wave
    // broadcasts (all 16 lanes same address).
    __shared__ float lds_xi[16 * 132];
    __shared__ float lds_xj[16 * 132];
    __shared__ float lds_a[F];

    const float* xb = x + (size_t)b * V * F;
    for (int e = lane; e < 16 * F; e += 32) {
        const int r = e >> 7, f = e & (F - 1);
        lds_xi[r * 132 + f] = xb[(size_t)(ti * 16 + r) * F + f];
        lds_xj[r * 132 + f] = xb[(size_t)(tj * 16 + r) * F + f];
    }
    for (int e = lane; e < F; e += 32) lds_a[e] = aw[e];
    __syncthreads();

    // ---- Gram tile via WMMA over K = 0..127 in chunks of 32 ----
    const _Float16* xhb = xh + (size_t)b * V * F;
    const _Float16* pa  = xhb + (size_t)(ti * 16 + lm) * F;  // A row (M = lm)
    const _Float16* pb  = xhb + (size_t)(tj * 16 + lm) * F;  // B col (N = lm)
    v8f c = {};
#pragma unroll
    for (int k0 = 0; k0 < F; k0 += 32) {
        union { v16h v; v8h h[2]; } A, B;
        A.h[0] = *(const v8h*)(pa + k0 + half * 16);       // K = h*16 + 0..7
        A.h[1] = *(const v8h*)(pa + k0 + half * 16 + 8);   // K = h*16 + 8..15
        B.h[0] = *(const v8h*)(pb + k0 + half * 16);       // K = h*16 + 0..7
        B.h[1] = *(const v8h*)(pb + k0 + half * 16 + 8);   // K = h*16 + 8..15
        c = __builtin_amdgcn_wmma_f32_16x16x32_f16(
                /*neg_a=*/false, A.v, /*neg_b=*/false, B.v,
                /*c_mod=*/(short)0, c, /*reuse_a=*/false, /*reuse_b=*/false);
    }

    // ---- weighted L1 tile (exact f32): lane owns column j, rows half*8+r ----
    float l1[8];
#pragma unroll
    for (int r = 0; r < 8; ++r) {
        const float* xi = &lds_xi[(half * 8 + r) * 132];   // broadcast per half
        const float* xj = &lds_xj[lm * 132];
        float acc = 0.f;
#pragma unroll 4
        for (int f = 0; f < F; ++f)
            acc = fmaf(fabsf(xi[f] - xj[f]), lds_a[f], acc);
        l1[r] = acc;
    }

    const int  j  = tj * 16 + lm;
    const float nj = nsq[b * V + j];
#pragma unroll
    for (int r = 0; r < 8; ++r) {
        const int   gi = ti * 16 + half * 8 + r;
        const float d2 = nsq[b * V + gi] + nj - 2.f * c[r];
        const float e  = __expf(-fmaxf(l1[r], 0.f));       // relu then exp(-.)
        const size_t o = ((size_t)b * V + gi) * V + j;
        E[o]  = e;
        D2[o] = d2;
    }
}

// ---------------------------------------------------------------------------
// Kernel 3: deterministic column sums (torch sums over dim=1, i.e. over i).
// One thread per (b, j); consecutive threads -> consecutive j -> coalesced.
// ---------------------------------------------------------------------------
__global__ void colsum_kernel(const float* __restrict__ E,
                              float* __restrict__ colsum) {
    const int t = blockIdx.x * blockDim.x + threadIdx.x;   // 0 .. NB*V-1
    const int b = t / V, j = t % V;
    const float* col = E + (size_t)b * VV + j;
    float s = 0.f;
    for (int i = 0; i < V; ++i) s += col[(size_t)i * V];
    colsum[t] = s;
}

// ---------------------------------------------------------------------------
// Kernel 4: s = E / colsum -> d_out ; per-block partial of (s^2 + alpha*D2*s).
// ---------------------------------------------------------------------------
__global__ void finalize_kernel(const float* __restrict__ E,
                                const float* __restrict__ D2,
                                const float* __restrict__ colsum,
                                const float* __restrict__ alpha_p,
                                float* __restrict__ s_out,
                                float* __restrict__ partial) {
    const size_t idx = (size_t)blockIdx.x * blockDim.x + threadIdx.x;
    const int b   = (int)(idx / VV);
    const int j   = (int)(idx % V);
    const float alpha = alpha_p[0];
    const float s = E[idx] / colsum[b * V + j];
    s_out[idx] = s;
    float tv = s * s + alpha * D2[idx] * s;
    __shared__ float red[256];
    red[threadIdx.x] = tv;
    __syncthreads();
    for (int st = 128; st > 0; st >>= 1) {
        if ((int)threadIdx.x < st) red[threadIdx.x] += red[threadIdx.x + st];
        __syncthreads();
    }
    if (threadIdx.x == 0) partial[blockIdx.x] = red[0];
}

// ---------------------------------------------------------------------------
// Kernel 5: fixed-order reduction of block partials -> scalar loss.
// ---------------------------------------------------------------------------
__global__ void reduce_kernel(const float* __restrict__ partial, int n,
                              float* __restrict__ out) {
    float s = 0.f;
    for (int i = threadIdx.x; i < n; i += 256) s += partial[i];
    __shared__ float red[256];
    red[threadIdx.x] = s;
    __syncthreads();
    for (int st = 128; st > 0; st >>= 1) {
        if ((int)threadIdx.x < st) red[threadIdx.x] += red[threadIdx.x + st];
        __syncthreads();
    }
    if (threadIdx.x == 0) out[0] = red[0];
}

// ---------------------------------------------------------------------------
// Launcher. Workspace layout (all offsets 256B-aligned, total ~8.95 MB):
//   xh      f16  NB*V*F      @ 0         (512 KB)
//   nsq     f32  NB*V        @ 524288    (  8 KB)
//   E       f32  NB*V*V      @ 532480    (  4 MB)
//   D2      f32  NB*V*V      @ 4726784   (  4 MB)
//   colsum  f32  NB*V        @ 8921088   (  8 KB)
//   partial f32  NB*VV/256   @ 8929280   ( 16 KB)
// ---------------------------------------------------------------------------
extern "C" void kernel_launch(void* const* d_in, const int* in_sizes, int n_in,
                              void* d_out, int out_size, void* d_ws, size_t ws_size,
                              hipStream_t stream) {
    (void)in_sizes; (void)n_in; (void)out_size; (void)ws_size;
    const float* x     = (const float*)d_in[0];   // [NB,V,F] f32
    const float* a     = (const float*)d_in[1];   // [F,1]    f32
    const float* alpha = (const float*)d_in[2];   // scalar   f32
    float* out = (float*)d_out;                   // s (NB*V*V) then ajloss (1)

    char* ws = (char*)d_ws;
    _Float16* xh   = (_Float16*)(ws);
    float* nsq     = (float*)(ws + 524288);
    float* E       = (float*)(ws + 532480);
    float* D2      = (float*)(ws + 4726784);
    float* colsum  = (float*)(ws + 8921088);
    float* partial = (float*)(ws + 8929280);

    prep_kernel<<<NB * V, F, 0, stream>>>(x, xh, nsq);
    tile_kernel<<<NB * 32 * 32, 32, 0, stream>>>(x, xh, a, nsq, E, D2);
    colsum_kernel<<<(NB * V) / 256, 256, 0, stream>>>(E, colsum);
    finalize_kernel<<<(NB * VV) / 256, 256, 0, stream>>>(E, D2, colsum, alpha,
                                                         out, partial);
    reduce_kernel<<<1, 256, 0, stream>>>(partial, (NB * VV) / 256,
                                         out + (size_t)NB * VV);
}
